// I2C_KNN_80015240724888
// MI455X (gfx1250) — matrix operand order repaired and checked by reading, not compiled, not run on hardware
//
#include <hip/hip_runtime.h>

// ---------------------------------------------------------------------------
// Problem constants (from the reference)
// ---------------------------------------------------------------------------
#define BATCH      32
#define HW         441           // 21*21
#define CHANNELS   64
#define NCLASS     5
#define PER_CLS    2205          // 5 support images * 441
#define QPAD       448           // 441 padded to multiple of 16
#define SPAD       2208          // 2205 padded to multiple of 16
#define QTILES     (QPAD / 16)   // 28
#define STILES     (SPAD / 16)   // 138
#define NWAVES     8

typedef __attribute__((ext_vector_type(16))) _Float16 v16h;
typedef __attribute__((ext_vector_type(8)))  _Float16 v8h;
typedef __attribute__((ext_vector_type(8)))  float    v8f;

// ---------------------------------------------------------------------------
// Kernel 1: NCHW fp32 -> row-major unit-L2 f16 rows, with zeroed pad rows.
//   Query:   groups=32 (batch), imgsPerGroup=1, V=441,  P=448
//   Support: groups=5  (class), imgsPerGroup=5, V=2205, P=2208
// Consecutive threads read consecutive hw -> fully coalesced input loads.
// ---------------------------------------------------------------------------
__global__ __launch_bounds__(256) void normalize_f16_kernel(
    const float* __restrict__ in, _Float16* __restrict__ out,
    int groups, int imgsPerGroup, int V, int P)
{
    int idx = blockIdx.x * blockDim.x + threadIdx.x;
    if (idx >= groups * P) return;
    int g = idx / P;
    int r = idx - g * P;
    _Float16* orow = out + (size_t)idx * CHANNELS;

    if (r >= V) {                 // pad row: write zeros (re-done every call)
        #pragma unroll
        for (int c = 0; c < CHANNELS; ++c) orow[c] = (_Float16)0.0f;
        return;
    }
    int j  = r / HW;
    int hw = r - j * HW;
    const float* base = in + ((size_t)(g * imgsPerGroup + j) * CHANNELS) * HW + hw;

    float vals[CHANNELS];
    float ss = 0.0f;
    #pragma unroll
    for (int c = 0; c < CHANNELS; ++c) {
        float v = base[(size_t)c * HW];
        vals[c] = v;
        ss += v * v;
    }
    float inv = rsqrtf(ss);
    #pragma unroll
    for (int c = 0; c < CHANNELS; ++c)
        orow[c] = (_Float16)(vals[c] * inv);
}

// ---------------------------------------------------------------------------
// Branchless ordered top-3 insert (t0 >= t1 >= t2 invariant).
// ---------------------------------------------------------------------------
__device__ __forceinline__ void top3_insert(float v, float& t0, float& t1, float& t2)
{
    float m0 = fmaxf(t0, v);
    float n0 = fminf(t0, v);
    float m1 = fmaxf(t1, n0);
    float n1 = fminf(t1, n0);
    t2 = fmaxf(t2, n1);
    t1 = m1;
    t0 = m0;
}

__device__ __forceinline__ float sigmoidf_dev(float x)
{
    return 1.0f / (1.0f + __expf(-x));
}

// ---------------------------------------------------------------------------
// Kernel 2: fused  sim = q . s^T  -> per-class top-3 -> sigmoid -> sum.
// One block per (batch, class).  Each wave register-blocks TWO 16-query
// tiles, so every A fragment (16 support descriptors x 64 ch) feeds 4 WMMAs:
//   accA = A0*bfA0 ; accB = A0*bfB0 ; accA += A1*bfA1 ; accB += A1*bfB1
// This halves support-descriptor (L0/L2) traffic vs 1 qtile/wave and keeps
// two independent accumulator chains in flight (no back-to-back D->C RAW).
//
// D layout: lane L<16 holds (M=r, N=L) in acc[r]; lane L+16 holds (M=8+r,
// N=L) -> each lane sees 8 support rows of ONE query column -> per-lane
// branchless top-3, merged with the partner lane via shfl_xor(16).
// Sigmoid applied only to the 3 survivors (monotonic => same selection).
// All control flow is wave-uniform => EXEC all-ones at every WMMA.
// ---------------------------------------------------------------------------
__global__ __launch_bounds__(32 * NWAVES) void i2c_knn_fused_kernel(
    const _Float16* __restrict__ q,   // (32*448, 64) f16, unit rows
    const _Float16* __restrict__ s,   // (5*2208, 64) f16, unit rows
    float* __restrict__ out)          // (32, 5)
{
    const int b    = blockIdx.x;          // batch
    const int cls  = blockIdx.y;          // class
    const int wave = threadIdx.x >> 5;
    const int lane = threadIdx.x & 31;
    const int hl   = lane >> 4;           // half-wave selector (0/1)
    const int l15  = lane & 15;

    __shared__ float partial[NWAVES];

    const float NEG = -__builtin_inff();
    const _Float16* sBase = s + (size_t)cls * SPAD * CHANNELS;
    const _Float16* qBase = q + (size_t)b * QPAD * CHANNELS;

    float waveAcc = 0.0f;

    // 28 query tiles -> 2 passes of (wave, wave+8) pairs per wave.
    #pragma unroll 1
    for (int qp = 0; qp < 2; ++qp) {
        const int  qtA    = wave + qp * 16;              // always < 28
        const int  qtBraw = qtA + 8;                     // may be >= 28
        const bool bValid = (qtBraw < QTILES);
        const int  qtB    = bValid ? qtBraw : (QTILES - 1);  // clamp in-bounds

        // ---- B fragments (queries), loaded once per pass ----
        // lane: column N = l15, K = k0 + hl*16 + {0..15} (contiguous halves)
        const _Float16* qrowA = qBase + ((size_t)(qtA * 16 + l15)) * CHANNELS;
        const _Float16* qrowB = qBase + ((size_t)(qtB * 16 + l15)) * CHANNELS;
        v16h bfA0 = *(const v16h*)(qrowA + hl * 16);        // K 0..31
        v16h bfA1 = *(const v16h*)(qrowA + 32 + hl * 16);   // K 32..63
        v16h bfB0 = *(const v16h*)(qrowB + hl * 16);
        v16h bfB1 = *(const v16h*)(qrowB + 32 + hl * 16);

        float a0 = NEG, a1 = NEG, a2 = NEG;   // per-lane top-3, qtile A
        float b0 = NEG, b1 = NEG, b2 = NEG;   // per-lane top-3, qtile B

        for (int st = 0; st < STILES; ++st) {
            // ---- A fragments (support): lane M = l15,
            //      halves[0..15]=K k0+hl*16+{0..15} (contiguous halves)
            const _Float16* srow = sBase + ((size_t)(st * 16 + l15)) * CHANNELS;
            union { v16h v; v8h h[2]; } A0, A1;
            A0.v = *(const v16h*)(srow +       hl * 16);
            A1.v = *(const v16h*)(srow + 32  + hl * 16);

            v8f accA = {};
            v8f accB = {};
            accA = __builtin_amdgcn_wmma_f32_16x16x32_f16(
                     false, A0.v, false, bfA0, (short)0, accA, false, false);
            accB = __builtin_amdgcn_wmma_f32_16x16x32_f16(
                     false, A0.v, false, bfB0, (short)0, accB, false, false);
            accA = __builtin_amdgcn_wmma_f32_16x16x32_f16(
                     false, A1.v, false, bfA1, (short)0, accA, false, false);
            accB = __builtin_amdgcn_wmma_f32_16x16x32_f16(
                     false, A1.v, false, bfB1, (short)0, accB, false, false);

            // ---- fused top-3 update; mask support pad rows with -inf ----
            int baseRow = st * 16 + hl * 8;
            #pragma unroll
            for (int r = 0; r < 8; ++r) {
                bool ok = (baseRow + r) < PER_CLS;
                float vA = ok ? accA[r] : NEG;
                float vB = ok ? accB[r] : NEG;
                top3_insert(vA, a0, a1, a2);
                top3_insert(vB, b0, b1, b2);
            }
        }

        // merge half-columns (lane L: rows M%16<8, lane L+16: the rest)
        top3_insert(__shfl_xor(a0, 16, 32), a0, a1, a2);
        top3_insert(__shfl_xor(a1, 16, 32), a0, a1, a2);
        top3_insert(__shfl_xor(a2, 16, 32), a0, a1, a2);
        top3_insert(__shfl_xor(b0, 16, 32), b0, b1, b2);
        top3_insert(__shfl_xor(b1, 16, 32), b0, b1, b2);
        top3_insert(__shfl_xor(b2, 16, 32), b0, b1, b2);

        float sumA = sigmoidf_dev(a0) + sigmoidf_dev(a1) + sigmoidf_dev(a2);
        float sumB = sigmoidf_dev(b0) + sigmoidf_dev(b1) + sigmoidf_dev(b2);

        // lanes 0..15 carry the merged query columns; mask query pad rows
        bool vA = (hl == 0) && (qtA * 16 + l15 < HW);
        bool vB = (hl == 0) && bValid && (qtB * 16 + l15 < HW);
        float contrib = (vA ? sumA : 0.0f) + (vB ? sumB : 0.0f);

        // deterministic full-wave xor reduction (masked lanes contribute 0)
        #pragma unroll
        for (int off = 1; off < 32; off <<= 1)
            contrib += __shfl_xor(contrib, off, 32);
        waveAcc += contrib;          // same value on every lane
    }

    if (lane == 0) partial[wave] = waveAcc;
    __syncthreads();

    if (threadIdx.x == 0) {
        float total = 0.0f;
        #pragma unroll
        for (int w = 0; w < NWAVES; ++w) total += partial[w];   // fixed order
        out[b * NCLASS + cls] = total;
    }
}

// ---------------------------------------------------------------------------
// Launcher
//   d_in[0] anchor      (32,64,21,21) f32
//   d_in[1] support_set (25,64,21,21) f32
//   d_in[2] av_num = 1, d_in[3] sav_num = 1 (unused)
//   d_out   (32,5) f32
//   d_ws    scratch: q_f16 (32*448*64) + s_f16 (5*2208*64)  ~3.2 MB
// ---------------------------------------------------------------------------
extern "C" void kernel_launch(void* const* d_in, const int* in_sizes, int n_in,
                              void* d_out, int out_size, void* d_ws, size_t ws_size,
                              hipStream_t stream)
{
    const float* anchor  = (const float*)d_in[0];
    const float* support = (const float*)d_in[1];
    float* out = (float*)d_out;

    _Float16* q_f16 = (_Float16*)d_ws;
    _Float16* s_f16 = q_f16 + (size_t)BATCH * QPAD * CHANNELS;

    {   // queries: 32 groups * 448 rows
        int total = BATCH * QPAD;
        normalize_f16_kernel<<<(total + 255) / 256, 256, 0, stream>>>(
            anchor, q_f16, BATCH, 1, HW, QPAD);
    }
    {   // support: 5 class groups * 2208 rows (5 images each)
        int total = NCLASS * SPAD;
        normalize_f16_kernel<<<(total + 255) / 256, 256, 0, stream>>>(
            support, s_f16, NCLASS, 5, PER_CLS, SPAD);
    }

    dim3 grid(BATCH, NCLASS);
    i2c_knn_fused_kernel<<<grid, 32 * NWAVES, 0, stream>>>(q_f16, s_f16, out);
}
